// MultiHeadAttention_4844723110009
// MI455X (gfx1250) — matrix operand order repaired and checked by reading, not compiled, not run on hardware
//
#include <hip/hip_runtime.h>

#define DM 1024
#define NH 16
#define DH 64
#define BB 2
#define SS 2048
#define MM (BB*SS)
#define SC_LD 2056  // 2048 + 8 pad -> conflict-reduced LDS softmax

typedef __attribute__((ext_vector_type(16))) __bf16 v16bf;
typedef __attribute__((ext_vector_type(8)))  float  v8f;

union Frag {
  v16bf f;
  uint4 u[2];
  __bf16 h[16];
};

#define WMMA_BF16(A,B,C) \
  __builtin_amdgcn_wmma_f32_16x16x32_bf16(false,(A),false,(B),(short)0,(C),false,false)

static __device__ __forceinline__ __bf16 tobf(float x) { return (__bf16)x; }

// ---------------------------------------------------------------------------
// Projection GEMM: y = X @ W^T + b ;  X [4096,1024] f32, W [1024,1024] f32
// out: bf16, head-split [B,H,S,DH] (vtrans=0) or [B,H,DH,S] (vtrans=1)
// 64x64 tile per block, 8 waves, K-step 64 (4 WMMAs per staging round).
// ---------------------------------------------------------------------------
__global__ __launch_bounds__(256) void proj_gemm_kernel(
    const float* __restrict__ X, const float* __restrict__ W,
    const float* __restrict__ bias, __bf16* __restrict__ out, int vtrans)
{
  __shared__ __align__(16) __bf16 As[64*64];
  __shared__ __align__(16) __bf16 Bs[64*64];
  const int tid  = threadIdx.x;
  const int m0   = blockIdx.x * 64;
  const int n0   = blockIdx.y * 64;
  const int wave = tid >> 5, lane = tid & 31;
  const int half = lane >> 4, r = lane & 15;
  const int wm   = wave & 3,  wn = wave >> 2;
  v8f acc0 = {}, acc1 = {};
  for (int kt = 0; kt < DM/64; ++kt) {
#pragma unroll
    for (int i = 0; i < 4; ++i) {
      int c = i*256 + tid;
      int row = c >> 4;            // 16 float4 per 64-col row
      int col = (c & 15) * 4;
      float4 xa = *(const float4*)(X + (size_t)(m0+row)*DM + kt*64 + col);
      union { __bf16 h[4]; uint2 u; } p;
      p.h[0]=tobf(xa.x); p.h[1]=tobf(xa.y); p.h[2]=tobf(xa.z); p.h[3]=tobf(xa.w);
      *(uint2*)(As + row*64 + col) = p.u;
      float4 wv = *(const float4*)(W + (size_t)(n0+row)*DM + kt*64 + col);
      p.h[0]=tobf(wv.x); p.h[1]=tobf(wv.y); p.h[2]=tobf(wv.z); p.h[3]=tobf(wv.w);
      *(uint2*)(Bs + row*64 + col) = p.u;
    }
    __syncthreads();
#pragma unroll
    for (int ks = 0; ks < 2; ++ks) {
      Frag a, b0, b1;
      const __bf16* ab = As + (wm*16 + r)*64 + ks*32;
      a.u[0]  = *(const uint4*)(ab + half*16);
      a.u[1]  = *(const uint4*)(ab + half*16 + 8);
      const __bf16* bb0 = Bs + (wn*32 + r)*64 + ks*32 + half*16;
      b0.u[0] = *(const uint4*)(bb0);
      b0.u[1] = *(const uint4*)(bb0 + 8);
      const __bf16* bb1 = Bs + (wn*32 + 16 + r)*64 + ks*32 + half*16;
      b1.u[0] = *(const uint4*)(bb1);
      b1.u[1] = *(const uint4*)(bb1 + 8);
      acc0 = WMMA_BF16(a.f, b0.f, acc0);
      acc1 = WMMA_BF16(a.f, b1.f, acc1);
    }
    __syncthreads();
  }
  v8f accs[2] = {acc0, acc1};
  const int m_base = m0 + wm*16 + half*8;
#pragma unroll
  for (int j = 0; j < 2; ++j) {
    int n = n0 + wn*32 + j*16 + r;
    float bv = bias[n];
    int h = n >> 6, d = n & (DH-1);
#pragma unroll
    for (int i = 0; i < 8; ++i) {
      int m = m_base + i;
      int b = m >> 11, s = m & (SS-1);
      float val = accs[j][i] + bv;
      size_t idx = vtrans ? ((size_t)((b*NH + h)*DH + d))*SS + s
                          : ((size_t)((b*NH + h)*SS + s))*DH + d;
      out[idx] = tobf(val);
    }
  }
}

// ---------------------------------------------------------------------------
// Dense GEMM: op = ctx(bf16) @ W^T + b -> f32
// ---------------------------------------------------------------------------
__global__ __launch_bounds__(256) void dense_gemm_kernel(
    const __bf16* __restrict__ Xb, const float* __restrict__ W,
    const float* __restrict__ bias, float* __restrict__ out)
{
  __shared__ __align__(16) __bf16 As[64*64];
  __shared__ __align__(16) __bf16 Bs[64*64];
  const int tid  = threadIdx.x;
  const int m0   = blockIdx.x * 64;
  const int n0   = blockIdx.y * 64;
  const int wave = tid >> 5, lane = tid & 31;
  const int half = lane >> 4, r = lane & 15;
  const int wm   = wave & 3,  wn = wave >> 2;
  v8f acc0 = {}, acc1 = {};
  for (int kt = 0; kt < DM/64; ++kt) {
#pragma unroll
    for (int i = 0; i < 2; ++i) {             // A: 4096 bf16 = 512 uint4
      int c = i*256 + tid;
      int row = c >> 3;
      int col = (c & 7) * 8;
      *(uint4*)(As + row*64 + col) =
          *(const uint4*)(Xb + (size_t)(m0+row)*DM + kt*64 + col);
    }
#pragma unroll
    for (int i = 0; i < 4; ++i) {             // B: f32 -> bf16
      int c = i*256 + tid;
      int row = c >> 4;
      int col = (c & 15) * 4;
      float4 wv = *(const float4*)(W + (size_t)(n0+row)*DM + kt*64 + col);
      union { __bf16 h[4]; uint2 u; } p;
      p.h[0]=tobf(wv.x); p.h[1]=tobf(wv.y); p.h[2]=tobf(wv.z); p.h[3]=tobf(wv.w);
      *(uint2*)(Bs + row*64 + col) = p.u;
    }
    __syncthreads();
#pragma unroll
    for (int ks = 0; ks < 2; ++ks) {
      Frag a, b0, b1;
      const __bf16* ab = As + (wm*16 + r)*64 + ks*32;
      a.u[0]  = *(const uint4*)(ab + half*16);
      a.u[1]  = *(const uint4*)(ab + half*16 + 8);
      const __bf16* bb0 = Bs + (wn*32 + r)*64 + ks*32 + half*16;
      b0.u[0] = *(const uint4*)(bb0);
      b0.u[1] = *(const uint4*)(bb0 + 8);
      const __bf16* bb1 = Bs + (wn*32 + 16 + r)*64 + ks*32 + half*16;
      b1.u[0] = *(const uint4*)(bb1);
      b1.u[1] = *(const uint4*)(bb1 + 8);
      acc0 = WMMA_BF16(a.f, b0.f, acc0);
      acc1 = WMMA_BF16(a.f, b1.f, acc1);
    }
    __syncthreads();
  }
  v8f accs[2] = {acc0, acc1};
  const int m_base = m0 + wm*16 + half*8;
#pragma unroll
  for (int j = 0; j < 2; ++j) {
    int n = n0 + wn*32 + j*16 + r;
    float bv = bias[n];
#pragma unroll
    for (int i = 0; i < 8; ++i) {
      int m = m_base + i;
      out[(size_t)m*DM + n] = accs[j][i] + bv;
    }
  }
}

// ---------------------------------------------------------------------------
// Attention: one block per (b*h, 32-row q strip). Scores strip lives in LDS.
// Phase 1: S = QK^T * 1/8 (WMMA bf16). Phase 2: float4 softmax + single
// coalesced attn write. Phase 3: ctx = P @ V with V pre-transposed [B,H,DH,S].
// ---------------------------------------------------------------------------
__global__ __launch_bounds__(256) void attention_kernel(
    const __bf16* __restrict__ Qh, const __bf16* __restrict__ Kh,
    const __bf16* __restrict__ VTh, float* __restrict__ attn_out,
    __bf16* __restrict__ ctxb)
{
  extern __shared__ __align__(16) float smem[];
  float* Sc      = smem;                 // [32][SC_LD]
  float* red     = smem + 32*SC_LD;      // [256]
  float* rowstat = red + 256;            // [32]

  const int tid  = threadIdx.x;
  const int bh   = blockIdx.x >> 6;      // 0..31  (b*16 + h)
  const int qb   = blockIdx.x & 63;      // 0..63
  const int q0   = qb * 32;
  const int wave = tid >> 5, lane = tid & 31;
  const int half = lane >> 4, r = lane & 15;

  const __bf16* Q  = Qh  + (size_t)bh * SS * DH;
  const __bf16* K  = Kh  + (size_t)bh * SS * DH;
  const __bf16* VT = VTh + (size_t)bh * DH * SS;

  // ---- Phase 1: scores ----
  {
    const int qsub = wave & 1;           // 16-row q subtile
    const int ksub = wave >> 1;          // 0..3 key subtile within 64
    const __bf16* qrow = Q + (size_t)(q0 + qsub*16 + r) * DH;
    Frag a0, a1;
    a0.u[0] = *(const uint4*)(qrow + half*16);
    a0.u[1] = *(const uint4*)(qrow + half*16 + 8);
    a1.u[0] = *(const uint4*)(qrow + 32 + half*16);
    a1.u[1] = *(const uint4*)(qrow + 32 + half*16 + 8);
    for (int kt = 0; kt < 32; ++kt) {
      int key0 = kt*64 + ksub*16;
      const __bf16* krow = K + (size_t)(key0 + r) * DH;
      Frag b0, b1;
      b0.u[0] = *(const uint4*)(krow + half*16);
      b0.u[1] = *(const uint4*)(krow + half*16 + 8);
      b1.u[0] = *(const uint4*)(krow + 32 + half*16);
      b1.u[1] = *(const uint4*)(krow + 32 + half*16 + 8);
      v8f acc = {};
      acc = WMMA_BF16(a0.f, b0.f, acc);
      acc = WMMA_BF16(a1.f, b1.f, acc);
      float* dst = Sc + (size_t)(qsub*16 + half*8)*SC_LD + key0 + r;
#pragma unroll
      for (int i = 0; i < 8; ++i) dst[i*SC_LD] = acc[i] * 0.125f;
    }
  }
  __syncthreads();

  // ---- Phase 2: float4 softmax + coalesced attn write ----
  {
    const int row = tid >> 3, j = tid & 7;   // 8 threads per q row
    float* Sr = Sc + row*SC_LD;
    float mx = -3.402823466e38f;
    for (int c = j*4; c < SS; c += 32) {
      float4 f = *(const float4*)(Sr + c);
      mx = fmaxf(mx, fmaxf(fmaxf(f.x, f.y), fmaxf(f.z, f.w)));
    }
    red[tid] = mx;
    __syncthreads();
    if (j == 0) {
      float m2 = red[row*8];
#pragma unroll
      for (int t = 1; t < 8; ++t) m2 = fmaxf(m2, red[row*8+t]);
      rowstat[row] = m2;
    }
    __syncthreads();
    const float mm = rowstat[row];
    float sum = 0.f;
    for (int c = j*4; c < SS; c += 32) {
      float4 f = *(const float4*)(Sr + c);
      f.x = __expf(f.x - mm); f.y = __expf(f.y - mm);
      f.z = __expf(f.z - mm); f.w = __expf(f.w - mm);
      *(float4*)(Sr + c) = f;
      sum += (f.x + f.y) + (f.z + f.w);
    }
    red[tid] = sum;
    __syncthreads();
    if (j == 0) {
      float s2 = 0.f;
#pragma unroll
      for (int t = 0; t < 8; ++t) s2 += red[row*8+t];
      rowstat[row] = 1.f / s2;
    }
    __syncthreads();
    const float inv = rowstat[row];
    float* arow = attn_out + (size_t)bh * SS * SS + (size_t)(q0 + row) * SS;
    for (int c = j*4; c < SS; c += 32) {
      float4 f = *(const float4*)(Sr + c);
      f.x *= inv; f.y *= inv; f.z *= inv; f.w *= inv;
      *(float4*)(Sr + c) = f;
      *(float4*)(arow + c) = f;
    }
  }
  __syncthreads();

  // ---- Phase 3: ctx = P @ V ----
  {
    const int qsub = wave & 1;
    const int dsub = wave >> 1;          // 0..3 -> 16 d-cols each
    v8f acc = {};
    for (int kt = 0; kt < 64; ++kt) {
      const float* pbase = Sc + (size_t)(qsub*16 + r)*SC_LD + kt*32 + half*8;
      float4 f0 = *(const float4*)(pbase);
      float4 f1 = *(const float4*)(pbase + 4);
      float4 f2 = *(const float4*)(pbase + 16);
      float4 f3 = *(const float4*)(pbase + 20);
      Frag a;
      a.h[0]=tobf(f0.x);  a.h[1]=tobf(f0.y);  a.h[2]=tobf(f0.z);  a.h[3]=tobf(f0.w);
      a.h[4]=tobf(f1.x);  a.h[5]=tobf(f1.y);  a.h[6]=tobf(f1.z);  a.h[7]=tobf(f1.w);
      a.h[8]=tobf(f2.x);  a.h[9]=tobf(f2.y);  a.h[10]=tobf(f2.z); a.h[11]=tobf(f2.w);
      a.h[12]=tobf(f3.x); a.h[13]=tobf(f3.y); a.h[14]=tobf(f3.z); a.h[15]=tobf(f3.w);
      const __bf16* vrow = VT + (size_t)(dsub*16 + r)*SS + kt*32 + half*8;
      Frag b;
      b.u[0] = *(const uint4*)(vrow);
      b.u[1] = *(const uint4*)(vrow + 16);
      acc = WMMA_BF16(a.f, b.f, acc);
    }
    const int bb = bh >> 4, hh = bh & (NH-1);
#pragma unroll
    for (int i = 0; i < 8; ++i) {
      int s   = q0 + qsub*16 + i + half*8;
      int col = hh*DH + dsub*16 + r;
      ctxb[((size_t)(bb*SS + s))*DM + col] = tobf(acc[i]);
    }
  }
}

// ---------------------------------------------------------------------------
extern "C" void kernel_launch(void* const* d_in, const int* in_sizes, int n_in,
                              void* d_out, int out_size, void* d_ws, size_t ws_size,
                              hipStream_t stream) {
  (void)in_sizes; (void)n_in; (void)out_size; (void)ws_size;
  const float* q    = (const float*)d_in[0];
  const float* k    = (const float*)d_in[1];
  const float* v    = (const float*)d_in[2];
  const float* wq_w = (const float*)d_in[3];
  const float* wq_b = (const float*)d_in[4];
  const float* wk_w = (const float*)d_in[5];
  const float* wk_b = (const float*)d_in[6];
  const float* wv_w = (const float*)d_in[7];
  const float* wv_b = (const float*)d_in[8];
  const float* dw   = (const float*)d_in[9];
  const float* db   = (const float*)d_in[10];

  float* op   = (float*)d_out;                       // [B,S,DM]
  float* attn = op + (size_t)BB*SS*DM;               // [B,H,S,S]

  const size_t NELEM = (size_t)BB*SS*DM;             // 4,194,304 bf16 each
  __bf16* qhb  = (__bf16*)d_ws;
  __bf16* khb  = qhb + NELEM;
  __bf16* vTb  = khb + NELEM;
  __bf16* ctxb = vTb + NELEM;

  dim3 ggrid(MM/64, DM/64);
  dim3 gblk(256);
  proj_gemm_kernel<<<ggrid, gblk, 0, stream>>>(q, wq_w, wq_b, qhb, 0);
  proj_gemm_kernel<<<ggrid, gblk, 0, stream>>>(k, wk_w, wk_b, khb, 0);
  proj_gemm_kernel<<<ggrid, gblk, 0, stream>>>(v, wv_w, wv_b, vTb, 1);

  size_t smem_bytes = (size_t)(32*SC_LD + 256 + 32) * sizeof(float); // ~258 KB
  attention_kernel<<<dim3(BB*NH*(SS/32)), gblk, smem_bytes, stream>>>(
      qhb, khb, vTb, attn, ctxb);

  dense_gemm_kernel<<<ggrid, gblk, 0, stream>>>(ctxb, dw, db, op);
}
